// GCN_19954418057934
// MI455X (gfx1250) — compile-verified, matrix operand order checked by reading, not run on hardware
//
#include <hip/hip_runtime.h>
#include <hip/hip_bf16.h>
#include <math.h>

// ---------------------------------------------------------------------------
// Problem constants (match reference)
// ---------------------------------------------------------------------------
#define N_NODES 46080
#define NPG     90
#define NGRAPH  512
#define NEDGE   737280
#define EPG     (NEDGE / NGRAPH)   // 1440 edges per graph (contiguous blocks)
#define IN_DIM  90
#define IN_PAD  96                 // pad K to multiple of 32 for WMMA
#define H       64
#define KTOP    70
#define POOLED  (KTOP * H)         // 4480, multiple of 32

typedef _Float16 half8_t  __attribute__((ext_vector_type(8)));
typedef _Float16 half16_t __attribute__((ext_vector_type(16)));
typedef float    float8_t __attribute__((ext_vector_type(8)));

// ---------------------------------------------------------------------------
// Degree / normalization
// ---------------------------------------------------------------------------
__global__ __launch_bounds__(256) void k_init_deg(float* __restrict__ deg, int n) {
    int i = blockIdx.x * blockDim.x + threadIdx.x;
    if (i < n) deg[i] = 1.0f;                     // self-loop weight
}

__global__ __launch_bounds__(256) void k_deg_accum(const int* __restrict__ dst,
                                                   const float* __restrict__ ew,
                                                   float* __restrict__ deg, int e) {
    int i = blockIdx.x * blockDim.x + threadIdx.x;
    if (i < e) atomicAdd(&deg[dst[i]], ew[i]);
}

__global__ __launch_bounds__(256) void k_dinv(const float* __restrict__ deg,
                                              float* __restrict__ dinv, int n) {
    int i = blockIdx.x * blockDim.x + threadIdx.x;
    if (i < n) {
        float d = deg[i];
        dinv[i] = (d > 0.0f) ? rsqrtf(fmaxf(d, 1e-12f)) : 0.0f;
    }
}

// ---------------------------------------------------------------------------
// f32 -> f16 conversion with column padding (row-major, rows x colsp)
// ---------------------------------------------------------------------------
__global__ __launch_bounds__(256) void k_cvt_pad_f16(const float* __restrict__ src,
                                                     _Float16* __restrict__ dst,
                                                     int cols, int colsp, long total) {
    long i = (long)blockIdx.x * blockDim.x + threadIdx.x;
    if (i >= total) return;
    int c = (int)(i % colsp);
    long r = i / colsp;
    dst[i] = (c < cols) ? (_Float16)src[r * cols + c] : (_Float16)0.0f;
}

// W[K][Ncols] (row-major) -> Wt[Ncols][Kp] f16, zero-padded K -> Kp
__global__ __launch_bounds__(256) void k_transpose_cvt_f16(const float* __restrict__ W,
                                                           _Float16* __restrict__ Wt,
                                                           int K, int Kp, int Ncols,
                                                           long total) {
    long i = (long)blockIdx.x * blockDim.x + threadIdx.x;
    if (i >= total) return;
    int k = (int)(i % Kp);
    int n = (int)(i / Kp);
    Wt[i] = (k < K) ? (_Float16)W[(long)k * Ncols + n] : (_Float16)0.0f;
}

// ---------------------------------------------------------------------------
// WMMA GEMM:  C[M x 64] (f32) = A[M x K] (f16, lda) @ Bt^T  (+ bias)
// Bt is stored transposed: Bt[n][k], ldb, so B fragments load contiguous K.
// One wave (32 lanes) computes one 16x16 tile; 4 waves/block cover N=64.
// M must be a multiple of 16, K a multiple of 32.
//
// Fragment layouts (CDNA5 ISA, 16-bit A 16x32 / B 32x16):
//   lane L:  row/col = L&15,  k-half = (L<16 ? 0 : 8)
//   VGPR0-3 <- K[kh .. kh+7]   (8 contiguous halves = one b128 load)
//   VGPR4-7 <- K[16+kh .. 16+kh+7]
// ---------------------------------------------------------------------------
__global__ __launch_bounds__(128)
void k_gemm_wmma_f16(const _Float16* __restrict__ A, const _Float16* __restrict__ Bt,
                     const float* __restrict__ bias, float* __restrict__ C,
                     int K, int lda, int ldb) {
    const int lane  = threadIdx.x;        // 0..31 (wave32)
    const int ntile = threadIdx.y;        // 0..3
    const int mtile = blockIdx.x;
    const int lr  = lane & 15;
    const int khi = (lane >> 4) << 3;     // 0 or 8

    const _Float16* arow = A  + (long)(mtile * 16 + lr) * lda;
    const _Float16* brow = Bt + (long)(ntile * 16 + lr) * ldb;

    float8_t acc = {};
    for (int k = 0; k < K; k += 32) {
        half8_t alo = *reinterpret_cast<const half8_t*>(arow + k + khi);
        half8_t ahi = *reinterpret_cast<const half8_t*>(arow + k + 16 + khi);
        half8_t blo = *reinterpret_cast<const half8_t*>(brow + k + khi);
        half8_t bhi = *reinterpret_cast<const half8_t*>(brow + k + 16 + khi);
        half16_t a = __builtin_shufflevector(alo, ahi, 0,1,2,3,4,5,6,7,8,9,10,11,12,13,14,15);
        half16_t b = __builtin_shufflevector(blo, bhi, 0,1,2,3,4,5,6,7,8,9,10,11,12,13,14,15);
        // D = A x B + C, f32 accumulate  ->  v_wmma_f32_16x16x32_f16
        acc = __builtin_amdgcn_wmma_f32_16x16x32_f16(false, a, false, b,
                                                     (short)0, acc, false, false);
    }

    const int col  = ntile * 16 + lr;
    const int row0 = mtile * 16 + khi;    // lanes 0-15 -> M 0..7, lanes 16-31 -> M 8..15
    const float bv = bias ? bias[col] : 0.0f;
#pragma unroll
    for (int r = 0; r < 8; ++r)
        C[(long)(row0 + r) * H + col] = acc[r] + bv;
}

// ---------------------------------------------------------------------------
// GCN aggregation, one workgroup per graph. Edge list is contiguous per graph.
// acc[d] = dinv[d]^2 * xw[d]  +  sum_e  dinv[s]*ew*dinv[d] * xw[s]
// Epilogue adds bias; optionally also writes relu(h) as f16 for the next GEMM.
// LDS accumulator: 90*64 f32 = 23 KB (<< 320 KB/WGP) -> ds_add_f32 atomics.
// ---------------------------------------------------------------------------
__global__ __launch_bounds__(256)
void k_gcn_aggregate(const float* __restrict__ xw, const float* __restrict__ dinv,
                     const int* __restrict__ src, const int* __restrict__ dst,
                     const float* __restrict__ ew, const float* __restrict__ bias,
                     float* __restrict__ out_f32, _Float16* __restrict__ out_relu_f16) {
    __shared__ float acc[NPG * H];
    const int g = blockIdx.x;
    const int t = threadIdx.x;
    const int nbase = g * NPG;

    for (int i = t; i < NPG * H; i += 256) {
        int n = i >> 6;
        float di = dinv[nbase + n];
        acc[i] = xw[(long)(nbase + n) * H + (i & 63)] * di * di;
    }
    __syncthreads();

    const int ebase = g * EPG;
    for (int e = t; e < EPG; e += 256) {
        int s = src[ebase + e];
        int d = dst[ebase + e];
        float w = dinv[s] * ew[ebase + e] * dinv[d];
        const float4* rowv = reinterpret_cast<const float4*>(xw + (long)s * H);
        float* drow = acc + (d - nbase) * H;
#pragma unroll
        for (int c4 = 0; c4 < H / 4; ++c4) {
            float4 v = rowv[c4];
            atomicAdd(drow + c4 * 4 + 0, v.x * w);
            atomicAdd(drow + c4 * 4 + 1, v.y * w);
            atomicAdd(drow + c4 * 4 + 2, v.z * w);
            atomicAdd(drow + c4 * 4 + 3, v.w * w);
        }
    }
    __syncthreads();

    for (int i = t; i < NPG * H; i += 256) {
        float v = acc[i] + bias[i & 63];
        out_f32[(long)nbase * H + i] = v;
        if (out_relu_f16)
            out_relu_f16[(long)nbase * H + i] = (_Float16)fmaxf(v, 0.0f);
    }
}

// ---------------------------------------------------------------------------
// global_sort_pool: per graph rank nodes by channel 63 (descending, stable by
// index — matches jnp.argsort(-v)), scatter top-K rows into f16 pooled matrix.
// Ranks 0..NPG-1 are a permutation, so pooled[g][0..K*H) is fully written.
// ---------------------------------------------------------------------------
__global__ __launch_bounds__(128)
void k_sortpool(const float* __restrict__ h2, _Float16* __restrict__ pooled) {
    __shared__ float val[NPG];
    const int g = blockIdx.x;
    const int t = threadIdx.x;
    if (t < NPG) val[t] = h2[((long)(g * NPG + t)) * H + (H - 1)];
    __syncthreads();
    if (t < NPG) {
        float v = val[t];
        int r = 0;
        for (int j = 0; j < NPG; ++j) {
            float u = val[j];
            r += (u > v) || (u == v && j < t);
        }
        if (r < KTOP) {
            const float* row = h2 + (long)(g * NPG + t) * H;
            _Float16* orow = pooled + (long)g * POOLED + r * H;
#pragma unroll 8
            for (int c = 0; c < H; ++c) orow[c] = (_Float16)row[c];
        }
    }
}

// ---------------------------------------------------------------------------
// Final: out[g] = sigmoid( z1[g,:] . lw2 + lb2 )
// ---------------------------------------------------------------------------
__global__ __launch_bounds__(256)
void k_final(const float* __restrict__ z1, const float* __restrict__ lw2,
             const float* __restrict__ lb2, float* __restrict__ out) {
    int g = blockIdx.x * blockDim.x + threadIdx.x;
    if (g >= NGRAPH) return;
    float s = lb2[0];
    const float* row = z1 + (long)g * H;
#pragma unroll 8
    for (int c = 0; c < H; ++c) s += row[c] * lw2[c];
    out[g] = 1.0f / (1.0f + expf(-s));
}

// ---------------------------------------------------------------------------
// Host launch
// ---------------------------------------------------------------------------
extern "C" void kernel_launch(void* const* d_in, const int* in_sizes, int n_in,
                              void* d_out, int out_size, void* d_ws, size_t ws_size,
                              hipStream_t stream) {
    (void)n_in; (void)out_size; (void)ws_size;
    // Inputs (setup_inputs order)
    const float* x   = (const float*)d_in[0];                 // [N,90]
    const int*   src = (const int*)d_in[1];                   // edge_index[0], [E]
    const int*   dst = src + NEDGE;                           // edge_index[1], [E]
    const float* ew  = (const float*)d_in[2];                 // [E]
    const float* W1  = (const float*)d_in[4];                 // [90,64]
    const float* b1  = (const float*)d_in[5];                 // [64]
    const float* W2  = (const float*)d_in[6];                 // [64,64]
    const float* b2  = (const float*)d_in[7];                 // [64]
    const float* lw1 = (const float*)d_in[8];                 // [4480,64]
    const float* lb1 = (const float*)d_in[9];                 // [64]
    const float* lw2 = (const float*)d_in[10];                // [64,1]
    const float* lb2 = (const float*)d_in[11];                // [1]

    // Outputs: (probs[512], x_train[46080*64]) concatenated
    float* out_probs = (float*)d_out;
    float* out_xtr   = out_probs + NGRAPH;

    // Workspace carve-up (256B aligned)
    char* ws = (char*)d_ws;
    size_t off = 0;
    auto carve = [&](size_t bytes) { char* p = ws + off; off += (bytes + 255) & ~(size_t)255; return p; };
    float*     deg   = (float*)    carve((size_t)N_NODES * 4);
    float*     dinv  = (float*)    carve((size_t)N_NODES * 4);
    _Float16*  xh    = (_Float16*) carve((size_t)N_NODES * IN_PAD * 2);   // reused for pooled/z1 later
    _Float16*  w1t   = (_Float16*) carve((size_t)H * IN_PAD * 2);
    _Float16*  w2t   = (_Float16*) carve((size_t)H * H * 2);
    _Float16*  lw1t  = (_Float16*) carve((size_t)H * POOLED * 2);
    float*     xw    = (float*)    carve((size_t)N_NODES * H * 4);        // gemm1 out, then gemm2 out
    _Float16*  hh    = (_Float16*) carve((size_t)N_NODES * H * 2);        // relu(h1) f16
    float*     h2b   = (float*)    carve((size_t)N_NODES * H * 4);        // h2
    // alias into xh region (dead after gemm1): pooled f16 [512,4480] + z1 f32 [512,64]
    _Float16*  pooled = (_Float16*)xh;                                    // 4.59 MB < 8.85 MB
    float*     z1     = (float*)((char*)xh + (((size_t)NGRAPH * POOLED * 2 + 255) & ~(size_t)255));

    const int T = 256;
    auto blocks = [](long n, int t) { return (unsigned)((n + t - 1) / t); };

    // 1) degree / normalization
    k_init_deg <<<blocks(N_NODES, T), T, 0, stream>>>(deg, N_NODES);
    k_deg_accum<<<blocks(NEDGE,   T), T, 0, stream>>>(dst, ew, deg, NEDGE);
    k_dinv     <<<blocks(N_NODES, T), T, 0, stream>>>(deg, dinv, N_NODES);

    // 2) f16 staging: x padded, weights transposed
    k_cvt_pad_f16      <<<blocks((long)N_NODES * IN_PAD, T), T, 0, stream>>>(x,  xh,  IN_DIM, IN_PAD, (long)N_NODES * IN_PAD);
    k_transpose_cvt_f16<<<blocks((long)H * IN_PAD,       T), T, 0, stream>>>(W1, w1t, IN_DIM, IN_PAD, H, (long)H * IN_PAD);
    k_transpose_cvt_f16<<<blocks((long)H * H,            T), T, 0, stream>>>(W2, w2t, H, H, H, (long)H * H);
    k_transpose_cvt_f16<<<blocks((long)H * POOLED,       T), T, 0, stream>>>(lw1, lw1t, POOLED, POOLED, H, (long)H * POOLED);

    dim3 gblk(32, 4);
    // 3) GEMM1: xw = xh @ W1   [46080x96]x[96x64]
    k_gemm_wmma_f16<<<N_NODES / 16, gblk, 0, stream>>>(xh, w1t, nullptr, xw, IN_PAD, IN_PAD, IN_PAD);
    // 4) conv1 aggregate -> h1 (x_train output) + relu f16
    k_gcn_aggregate<<<NGRAPH, T, 0, stream>>>(xw, dinv, src, dst, ew, b1, out_xtr, hh);
    // 5) GEMM2: xw = hh @ W2   [46080x64]x[64x64]
    k_gemm_wmma_f16<<<N_NODES / 16, gblk, 0, stream>>>(hh, w2t, nullptr, xw, H, H, H);
    // 6) conv2 aggregate -> h2
    k_gcn_aggregate<<<NGRAPH, T, 0, stream>>>(xw, dinv, src, dst, ew, b2, h2b, nullptr);
    // 7) sort-pool -> pooled f16 [512,4480]
    k_sortpool<<<NGRAPH, 128, 0, stream>>>(h2b, pooled);
    // 8) GEMM3: z1 = pooled @ lw1 + lb1   [512x4480]x[4480x64]
    k_gemm_wmma_f16<<<NGRAPH / 16, gblk, 0, stream>>>(pooled, lw1t, lb1, z1, POOLED, POOLED, POOLED);
    // 9) final dot + sigmoid
    k_final<<<blocks(NGRAPH, T), T, 0, stream>>>(z1, lw2, lb2, out_probs);
}
